// LlamaSdpaAttention_72112500900442
// MI455X (gfx1250) — matrix operand order repaired and checked by reading, not compiled, not run on hardware
//
#include <hip/hip_runtime.h>

typedef __attribute__((ext_vector_type(16))) _Float16 v16h;
typedef __attribute__((ext_vector_type(8)))  _Float16 v8h;
typedef __attribute__((ext_vector_type(4)))  _Float16 v4h;
typedef __attribute__((ext_vector_type(8)))  float    v8f;
typedef __attribute__((ext_vector_type(4)))  float    v4f;

#define WMMA_F16(a, b, c) \
  __builtin_amdgcn_wmma_f32_16x16x32_f16(false, (a), false, (b), (short)0, (c), false, false)

static __device__ __forceinline__ v16h cat8(v8h lo, v8h hi) {
  return __builtin_shufflevector(lo, hi, 0, 1, 2, 3, 4, 5, 6, 7, 8, 9, 10, 11, 12, 13, 14, 15);
}

// Async global->LDS DMA availability (gfx1250 ASYNCcnt path)
#ifdef __has_builtin
#if __has_builtin(__builtin_amdgcn_global_load_async_to_lds_b128)
#define HAVE_ASYNC_LDS 1
#endif
#endif
#ifndef HAVE_ASYNC_LDS
#define HAVE_ASYNC_LDS 0
#endif

template <int N>
static __device__ __forceinline__ void wait_async() {
#if __has_builtin(__builtin_amdgcn_s_wait_asynccnt)
  __builtin_amdgcn_s_wait_asynccnt(N);
#else
  asm volatile("s_wait_asynccnt %0" ::"i"(N) : "memory");
#endif
}

// ---------------------------------------------------------------------------
// fp32 -> fp16 conversion, 4-wide (b128 loads / b64 stores)
// ---------------------------------------------------------------------------
__global__ void __launch_bounds__(256) cvt_f32_f16_x4(const v4f* __restrict__ src,
                                                      v4h* __restrict__ dst, int n4) {
  int i = blockIdx.x * blockDim.x + threadIdx.x;
  if (i < n4) dst[i] = __builtin_convertvector(src[i], v4h);
}

// ---------------------------------------------------------------------------
// GEMM: C[M,N] = A[M,K] @ W[N,K]^T  (f16 inputs, f32 WMMA accumulate)
// One wave computes a 16x64 tile of C.  Block = 4 waves = 64 rows x 64 cols.
// Direct-global-load variant (fallback, only instantiated without async DMA).
// ---------------------------------------------------------------------------
template <bool F32OUT>
__global__ void __launch_bounds__(128) gemm_xwT(const _Float16* __restrict__ A,
                                                const _Float16* __restrict__ W,
                                                void* __restrict__ Cout,
                                                int M, int N, int K) {
  const int lane = threadIdx.x & 31;
  const int wv   = threadIdx.x >> 5;
  const int lm   = lane & 15;
  const int lh   = lane >> 4;
  const int r0   = blockIdx.y * 64 + wv * 16;
  const int c0   = blockIdx.x * 64;

  const _Float16* Arow = A + (size_t)(r0 + lm) * K;
  v8f acc0 = {}, acc1 = {}, acc2 = {}, acc3 = {};

  for (int k0 = 0; k0 < K; k0 += 32) {
    v16h af = cat8(*(const v8h*)(Arow + k0 + lh * 16),
                   *(const v8h*)(Arow + k0 + 8 + lh * 16));
    const _Float16* Wb = W + (size_t)(c0 + lm) * K + k0 + lh * 16;
    if (k0 + 32 < K) __builtin_prefetch((const void*)(Wb + 32), 0, 0);
    v16h b0 = *(const v16h*)(Wb);
    v16h b1 = *(const v16h*)(Wb + (size_t)16 * K);
    v16h b2 = *(const v16h*)(Wb + (size_t)32 * K);
    v16h b3 = *(const v16h*)(Wb + (size_t)48 * K);
    acc0 = WMMA_F16(af, b0, acc0);
    acc1 = WMMA_F16(af, b1, acc1);
    acc2 = WMMA_F16(af, b2, acc2);
    acc3 = WMMA_F16(af, b3, acc3);
  }

#pragma unroll
  for (int v = 0; v < 8; ++v) {
    int row = r0 + v + lh * 8;
    size_t base = (size_t)row * N + c0 + lm;
    if (F32OUT) {
      float* C = (float*)Cout;
      C[base] = acc0[v]; C[base + 16] = acc1[v];
      C[base + 32] = acc2[v]; C[base + 48] = acc3[v];
    } else {
      _Float16* C = (_Float16*)Cout;
      C[base] = (_Float16)acc0[v]; C[base + 16] = (_Float16)acc1[v];
      C[base + 32] = (_Float16)acc2[v]; C[base + 48] = (_Float16)acc3[v];
    }
  }
}

#if HAVE_ASYNC_LDS
// ---------------------------------------------------------------------------
// Async-staged GEMM: the shared 64x32-half B tile is DMA'd into LDS once per
// block (4x less B traffic) via GLOBAL_LOAD_ASYNC_TO_LDS_B128 with a depth-4
// circular buffer: up to 3 stages in flight, one barrier per k-step
// (__syncthreads' s_wait_dscnt makes the prior iteration's LDS reads complete
// before any wave can re-stage that buffer).
// ---------------------------------------------------------------------------
typedef int v4i __attribute__((vector_size(16)));
typedef __attribute__((address_space(1))) v4i gv4i;
typedef __attribute__((address_space(3))) v4i lv4i;

template <bool F32OUT>
__global__ void __launch_bounds__(128) gemm_xwT_async(const _Float16* __restrict__ A,
                                                      const _Float16* __restrict__ W,
                                                      void* __restrict__ Cout,
                                                      int M, int N, int K) {
  constexpr int D = 4;                       // pipeline depth
  __shared__ __align__(32) _Float16 bstage[D][64 * 32];
  const int tid  = threadIdx.x;
  const int lane = tid & 31;
  const int wv   = tid >> 5;
  const int lm   = lane & 15;
  const int lh   = lane >> 4;
  const int r0   = blockIdx.y * 64 + wv * 16;
  const int c0   = blockIdx.x * 64;

  const _Float16* Arow = A + (size_t)(r0 + lm) * K;
  // staging map: thread t copies 32B of W row (t>>1), half-part (t&1)
  const _Float16* Wsrc = W + (size_t)(c0 + (tid >> 1)) * K + (tid & 1) * 16;
  const int ldsoff = (tid >> 1) * 32 + (tid & 1) * 16;

  auto stage = [&](int b, int k0) {
    gv4i* s = (gv4i*)(Wsrc + k0);
    lv4i* d = (lv4i*)(&bstage[b][ldsoff]);
    __builtin_amdgcn_global_load_async_to_lds_b128(s, d, 0, 0);
    __builtin_amdgcn_global_load_async_to_lds_b128(s + 1, d + 1, 0, 0);
  };

  v8f acc0 = {}, acc1 = {}, acc2 = {}, acc3 = {};
  const int nIt  = K / 32;
  const int npre = (nIt < D - 1) ? nIt : (D - 1);
  for (int d = 0; d < npre; ++d) stage(d, d * 32);   // prime the pipeline

  for (int it = 0; it < nIt; ++it) {
    const int k0 = it * 32;
    const int b  = it & (D - 1);
    // Ensure stage(it) landed: with a full pipe, leaving 2*(D-2) outstanding
    // retires the oldest stage; near the tail just drain.
    if (it + D - 2 < nIt) wait_async<2 * (D - 2)>();
    else                  wait_async<0>();
    __syncthreads();                    // buffer b visible to all 4 waves
    const _Float16* Bb = &bstage[b][0];
    v16h b0 = *(const v16h*)(Bb + lm * 32 + lh * 16);
    v16h b1 = *(const v16h*)(Bb + (16 + lm) * 32 + lh * 16);
    v16h b2 = *(const v16h*)(Bb + (32 + lm) * 32 + lh * 16);
    v16h b3 = *(const v16h*)(Bb + (48 + lm) * 32 + lh * 16);
    // Re-stage the oldest buffer ((it-1) % D): its readers all completed
    // before this iteration's barrier (barrier waits on DScnt).
    if (it + D - 1 < nIt) stage((it + D - 1) & (D - 1), k0 + (D - 1) * 32);
    v16h af = cat8(*(const v8h*)(Arow + k0 + lh * 16),
                   *(const v8h*)(Arow + k0 + 8 + lh * 16));
    acc0 = WMMA_F16(af, b0, acc0);
    acc1 = WMMA_F16(af, b1, acc1);
    acc2 = WMMA_F16(af, b2, acc2);
    acc3 = WMMA_F16(af, b3, acc3);
  }

#pragma unroll
  for (int v = 0; v < 8; ++v) {
    int row = r0 + v + lh * 8;
    size_t base = (size_t)row * N + c0 + lm;
    if (F32OUT) {
      float* C = (float*)Cout;
      C[base] = acc0[v]; C[base + 16] = acc1[v];
      C[base + 32] = acc2[v]; C[base + 48] = acc3[v];
    } else {
      _Float16* C = (_Float16*)Cout;
      C[base] = (_Float16)acc0[v]; C[base + 16] = (_Float16)acc1[v];
      C[base + 32] = (_Float16)acc2[v]; C[base + 48] = (_Float16)acc3[v];
    }
  }
}
#define GEMM_KERNEL gemm_xwT_async
#else
#define GEMM_KERNEL gemm_xwT
#endif

// ---------------------------------------------------------------------------
// RoPE (reference's interleaved-index variant) + relayout to (b,h,s,d).
// ---------------------------------------------------------------------------
static __device__ __forceinline__ float rope_invf(int t) {
  return __powf(10000.0f, -(float)t * (1.0f / 32.0f));
}
static __device__ __forceinline__ float rope_emb(int idx, float pos) {
  return (idx < 32) ? sinf(pos * rope_invf(idx)) : cosf(pos * rope_invf(idx - 32));
}

__global__ void __launch_bounds__(256) rope_qk(const _Float16* __restrict__ in,
                                               _Float16* __restrict__ out, float scale) {
  int i = blockIdx.x * blockDim.x + threadIdx.x;  // B*S*NH*32 = 2M threads
  int j = i & 31;
  int h = (i >> 5) & 15;
  int s = (i >> 9) & 2047;
  int b = i >> 20;
  float pos = (float)s;
  float sin_r = rope_emb(2 * j, pos);
  float cos_r = rope_emb(2 * j + 1, pos);
  const _Float16* x = in + ((size_t)(b * 2048 + s)) * 1024 + h * 64;
  float xe = (float)x[2 * j];
  float xo = (float)x[2 * j + 1];
  _Float16* o = out + (((size_t)(b * 16 + h) * 2048) + s) * 64;
  o[j]      = (_Float16)((xe * cos_r - xo * sin_r) * scale);
  o[32 + j] = (_Float16)((xe * sin_r + xo * cos_r) * scale);
}

// ---------------------------------------------------------------------------
// V relayout: (b,s,h*64+d) -> (b,h,d,s)
// ---------------------------------------------------------------------------
__global__ void __launch_bounds__(256) v_transpose(const _Float16* __restrict__ in,
                                                   _Float16* __restrict__ out) {
  int i = blockIdx.x * blockDim.x + threadIdx.x;  // B*NH*HD*S = 4M
  int s = i & 2047;
  int d = (i >> 11) & 63;
  int h = (i >> 17) & 15;
  int b = i >> 21;
  out[i] = in[((size_t)(b * 2048 + s)) * 1024 + h * 64 + d];
}

// ---------------------------------------------------------------------------
// Flash attention, transposed-score formulation:
// query on the lane axis -> per-lane softmax, vectorized mask loads, and the
// exp() outputs land directly in the A-fragment layout for P @ V.
// ---------------------------------------------------------------------------
__global__ void __launch_bounds__(128) flash_attn(const _Float16* __restrict__ Q,
                                                  const _Float16* __restrict__ Kd,
                                                  const _Float16* __restrict__ Vt,
                                                  const float* __restrict__ mask,
                                                  _Float16* __restrict__ O) {
  const int lane = threadIdx.x & 31;
  const int wv   = threadIdx.x >> 5;
  const int lm   = lane & 15;
  const int lh   = lane >> 4;

  int g  = blockIdx.x * 4 + wv;   // 4096 waves total
  int qt = g & 127;
  int bh = g >> 7;                // 0..31
  const _Float16* Qb = Q  + (size_t)bh * 2048 * 64;
  const _Float16* Kb = Kd + (size_t)bh * 2048 * 64;
  const _Float16* Vb = Vt + (size_t)bh * 64 * 2048;
  const int qbase = qt * 16;

  const _Float16* Qr = Qb + (size_t)(qbase + lm) * 64 + lh * 16;
  const v16h qb0 = *(const v16h*)(Qr);
  const v16h qb1 = *(const v16h*)(Qr + 32);

  const float* mrow = mask + (size_t)(qbase + lm) * 2048 + 8 * lh;

  float mi = -3.0e38f, li = 0.0f;
  v8f o0 = {}, o1 = {}, o2 = {}, o3 = {};

  const int kend = qbase + 16;      // causal: later key blocks fully masked
  for (int kc = 0; kc < kend; kc += 32) {
    const _Float16* K0 = Kb + (size_t)(kc + lm) * 64;
    const _Float16* K1 = K0 + 16 * 64;
    v8f s0 = {}, s1 = {};
    v16h ka;
    ka = cat8(*(const v8h*)(K0 + lh * 16), *(const v8h*)(K0 + 8 + lh * 16));
    s0 = WMMA_F16(ka, qb0, s0);
    ka = cat8(*(const v8h*)(K0 + 32 + lh * 16), *(const v8h*)(K0 + 40 + lh * 16));
    s0 = WMMA_F16(ka, qb1, s0);
    ka = cat8(*(const v8h*)(K1 + lh * 16), *(const v8h*)(K1 + 8 + lh * 16));
    s1 = WMMA_F16(ka, qb0, s1);
    ka = cat8(*(const v8h*)(K1 + 32 + lh * 16), *(const v8h*)(K1 + 40 + lh * 16));
    s1 = WMMA_F16(ka, qb1, s1);

    s0 += *(const v8f*)(mrow + kc);
    s1 += *(const v8f*)(mrow + kc + 16);

    float t = -3.0e38f;
#pragma unroll
    for (int v = 0; v < 8; ++v) t = fmaxf(t, fmaxf(s0[v], s1[v]));
    t = fmaxf(t, __shfl_xor(t, 16, 32));
    float mn = fmaxf(mi, t);
    float alpha = __expf(mi - mn);
    mi = mn;
    v8f p0, p1;
    float rs = 0.0f;
#pragma unroll
    for (int v = 0; v < 8; ++v) {
      p0[v] = __expf(s0[v] - mn);
      p1[v] = __expf(s1[v] - mn);
      rs += p0[v] + p1[v];
    }
    rs += __shfl_xor(rs, 16, 32);
    li = li * alpha + rs;

#pragma unroll
    for (int v = 0; v < 8; ++v) {
      float av = __shfl(alpha, v + 8 * lh, 32);
      o0[v] *= av; o1[v] *= av; o2[v] *= av; o3[v] *= av;
    }

    v16h pf = cat8(__builtin_convertvector(p0, v8h),
                   __builtin_convertvector(p1, v8h));

    const _Float16* V0 = Vb + (size_t)lm * 2048 + kc + lh * 8;
    o0 = WMMA_F16(pf, cat8(*(const v8h*)(V0), *(const v8h*)(V0 + 16)), o0);
    o1 = WMMA_F16(pf, cat8(*(const v8h*)(V0 + (size_t)16 * 2048), *(const v8h*)(V0 + (size_t)16 * 2048 + 16)), o1);
    o2 = WMMA_F16(pf, cat8(*(const v8h*)(V0 + (size_t)32 * 2048), *(const v8h*)(V0 + (size_t)32 * 2048 + 16)), o2);
    o3 = WMMA_F16(pf, cat8(*(const v8h*)(V0 + (size_t)48 * 2048), *(const v8h*)(V0 + (size_t)48 * 2048 + 16)), o3);
  }

  int b = bh >> 4, h = bh & 15;
#pragma unroll
  for (int v = 0; v < 8; ++v) {
    float lv = __shfl(li, v + 8 * lh, 32);
    float r = 1.0f / lv;
    size_t base = ((size_t)(b * 2048 + qbase + v + lh * 8)) * 1024 + h * 64 + lm;
    O[base]      = (_Float16)(o0[v] * r);
    O[base + 16] = (_Float16)(o1[v] * r);
    O[base + 32] = (_Float16)(o2[v] * r);
    O[base + 48] = (_Float16)(o3[v] * r);
  }
}

// ---------------------------------------------------------------------------
// Host-side orchestration
// ---------------------------------------------------------------------------
extern "C" void kernel_launch(void* const* d_in, const int* in_sizes, int n_in,
                              void* d_out, int out_size, void* d_ws, size_t ws_size,
                              hipStream_t stream) {
  (void)in_sizes; (void)n_in; (void)out_size; (void)ws_size;
  const float* hs   = (const float*)d_in[0];  // (2,2048,1024)
  const float* mask = (const float*)d_in[1];  // (1,1,2048,2048)
  const float* wq   = (const float*)d_in[2];
  const float* wk   = (const float*)d_in[3];
  const float* wv   = (const float*)d_in[4];
  const float* wo   = (const float*)d_in[5];

  const int M = 4096, HID = 1024;
  const size_t NX = (size_t)M * HID;       // 4,194,304
  const size_t NW = (size_t)HID * HID;     // 1,048,576

  char* ws = (char*)d_ws;
  size_t off = 0;
  auto alloc = [&](size_t halves) { void* q = ws + off; off += halves * sizeof(_Float16); return q; };
  _Float16* Xh   = (_Float16*)alloc(NX);
  _Float16* Wqh  = (_Float16*)alloc(NW);
  _Float16* Wkh  = (_Float16*)alloc(NW);
  _Float16* Wvh  = (_Float16*)alloc(NW);
  _Float16* Woh  = (_Float16*)alloc(NW);
  _Float16* Qlin = (_Float16*)alloc(NX);
  _Float16* Klin = (_Float16*)alloc(NX);
  _Float16* Vlin = (_Float16*)alloc(NX);
  _Float16* Kh   = (_Float16*)alloc(NX);
  _Float16* Vth  = (_Float16*)alloc(NX);
  _Float16* Qh   = Xh;    // X dead after projections
  _Float16* AOh  = Qlin;  // Qlin dead after RoPE(Q)

  cvt_f32_f16_x4<<<(int)(NX / 4 / 256), 256, 0, stream>>>((const v4f*)hs, (v4h*)Xh, (int)(NX / 4));
  cvt_f32_f16_x4<<<(int)(NW / 4 / 256), 256, 0, stream>>>((const v4f*)wq, (v4h*)Wqh, (int)(NW / 4));
  cvt_f32_f16_x4<<<(int)(NW / 4 / 256), 256, 0, stream>>>((const v4f*)wk, (v4h*)Wkh, (int)(NW / 4));
  cvt_f32_f16_x4<<<(int)(NW / 4 / 256), 256, 0, stream>>>((const v4f*)wv, (v4h*)Wvh, (int)(NW / 4));
  cvt_f32_f16_x4<<<(int)(NW / 4 / 256), 256, 0, stream>>>((const v4f*)wo, (v4h*)Woh, (int)(NW / 4));

  dim3 gg(HID / 64, M / 64), gb(128);
  GEMM_KERNEL<false><<<gg, gb, 0, stream>>>(Xh, Wqh, Qlin, M, HID, HID);
  GEMM_KERNEL<false><<<gg, gb, 0, stream>>>(Xh, Wkh, Klin, M, HID, HID);
  GEMM_KERNEL<false><<<gg, gb, 0, stream>>>(Xh, Wvh, Vlin, M, HID, HID);

  rope_qk<<<2097152 / 256, 256, 0, stream>>>(Qlin, Qh, 0.125f);
  rope_qk<<<2097152 / 256, 256, 0, stream>>>(Klin, Kh, 1.0f);

  v_transpose<<<(int)(NX / 256), 256, 0, stream>>>(Vlin, Vth);

  flash_attn<<<1024, 128, 0, stream>>>(Qh, Kh, Vth, mask, AOh);

  GEMM_KERNEL<true><<<gg, gb, 0, stream>>>(AOh, Woh, d_out, M, HID, HID);
}
